// Decoder_42743514530404
// MI455X (gfx1250) — compile-verified, matrix operand order checked by reading, not run on hardware
//
#include <hip/hip_runtime.h>
#include <hip/hip_bf16.h>

// ---------------- problem constants ----------------
#define Bsz   2048
#define DIMv  256
#define HIDv  512
#define MAXNv 128
#define MIDS  256   // (HID+1)//2
#define MIDK  320   // (MAXN+HID)//2
#define MIDD  384   // (HID+DIM)//2

typedef __attribute__((ext_vector_type(16))) __bf16 v16bf;
typedef __attribute__((ext_vector_type(8)))  __bf16 v8bf;
typedef __attribute__((ext_vector_type(8)))  float  v8f;

// Branch-free mish: tanh(softplus(x)) == t(t+2)/(t(t+2)+2) with t = e^x.
// Clamp exp input at 20 (ratio is 1.0 to f32 precision beyond that).
__device__ __forceinline__ float mishf(float x) {
    float t = __expf(fminf(x, 20.f));
    float u = t * (t + 2.f);
    return x * u * __builtin_amdgcn_rcpf(u + 2.f);
}

// ---------------- keys = mish(LN(kn_W1 + b1)) @ kn_W2 + b2 ; pos = I so pos@W1 = W1
__global__ __launch_bounds__(512) void k_keys(
    const float* __restrict__ W1, const float* __restrict__ b1,
    const float* __restrict__ g,  const float* __restrict__ be,
    const float* __restrict__ W2, const float* __restrict__ b2,
    float* __restrict__ keys)
{
    __shared__ float mid[MIDK];
    __shared__ float red[512];
    __shared__ float red2[512];
    int tid = threadIdx.x, row = blockIdx.x;
    float acc = 0.f;
    if (tid < MIDK) acc = W1[row * MIDK + tid] + b1[tid];
    red[tid]  = (tid < MIDK) ? acc : 0.f;
    red2[tid] = (tid < MIDK) ? acc * acc : 0.f;
    __syncthreads();
    for (int s = 256; s > 0; s >>= 1) {
        if (tid < s) { red[tid] += red[tid + s]; red2[tid] += red2[tid + s]; }
        __syncthreads();
    }
    float mean = red[0] * (1.f / MIDK);
    float var  = red2[0] * (1.f / MIDK) - mean * mean;
    __syncthreads();
    if (tid < MIDK)
        mid[tid] = mishf((acc - mean) * rsqrtf(var + 1e-5f) * g[tid] + be[tid]);
    __syncthreads();
    float o = b2[tid];
    #pragma unroll 8
    for (int j = 0; j < MIDK; j++) o += mid[j] * W2[j * HIDv + tid];
    keys[row * HIDv + tid] = o;
}

// ---------------- swizzle f32 weights into bf16 WMMA B-fragment order
// frag(kt,nt) -> 32 lanes x 16 halves contiguous.
// lane l holds column n = nt*16 + (l&15); halves j: K = kt*32 + (j<8 ? base+j : 16+base+j-8),
// base = (l<16 ? 0 : 8)  (mirrors the 16-bit A-matrix ISA layout).
__global__ void k_swizzle(const float* __restrict__ W, __bf16* __restrict__ Wf,
                          int NT, int Ncols, int total)
{
    int idx = blockIdx.x * blockDim.x + threadIdx.x;
    if (idx >= total) return;
    int j    = idx & 15;
    int lane = (idx >> 4) & 31;
    int frag = idx >> 9;
    int nt   = frag % NT;
    int kt   = frag / NT;
    int n    = nt * 16 + (lane & 15);
    int base = (lane < 16) ? 0 : 8;
    int kk   = kt * 32 + ((j < 8) ? (base + j) : (16 + base + (j - 8)));
    Wf[idx]  = (__bf16)W[kk * Ncols + n];
}

// ---------------- size_pred: n = clip(round(mish(LN(z@W1+b))@W2 + b2), 0, 128)
__global__ __launch_bounds__(256) void k_size_pred(
    const float* __restrict__ z,  const float* __restrict__ W1, const float* __restrict__ b1,
    const float* __restrict__ g,  const float* __restrict__ be,
    const float* __restrict__ W2, const float* __restrict__ b2,
    int* __restrict__ n_i, float* __restrict__ n_f, float* __restrict__ mask_out)
{
    __shared__ float zr[HIDv];
    __shared__ float red[MIDS];
    __shared__ float red2[MIDS];
    __shared__ int   nsh;
    int tid = threadIdx.x, row = blockIdx.x;
    zr[tid]       = z[row * HIDv + tid];
    zr[tid + 256] = z[row * HIDv + tid + 256];
    __syncthreads();
    float acc = b1[tid];
    #pragma unroll 8
    for (int k = 0; k < HIDv; k++) acc += zr[k] * W1[k * MIDS + tid];
    red[tid] = acc; red2[tid] = acc * acc;
    __syncthreads();
    for (int s = 128; s > 0; s >>= 1) {
        if (tid < s) { red[tid] += red[tid + s]; red2[tid] += red2[tid + s]; }
        __syncthreads();
    }
    float mean = red[0] * (1.f / MIDS);
    float var  = red2[0] * (1.f / MIDS) - mean * mean;
    __syncthreads();
    float hm = mishf((acc - mean) * rsqrtf(var + 1e-5f) * g[tid] + be[tid]);
    red[tid] = hm * W2[tid];
    __syncthreads();
    for (int s = 128; s > 0; s >>= 1) {
        if (tid < s) red[tid] += red[tid + s];
        __syncthreads();
    }
    if (tid == 0) {
        float logit = red[0] + b2[0];
        int nn = (int)fminf(fmaxf(rintf(logit), 0.f), (float)MAXNv);
        nsh = nn; n_i[row] = nn; n_f[row] = (float)nn;
    }
    __syncthreads();
    if (tid < MAXNv) mask_out[row * MAXNv + tid] = (tid < nsh) ? 1.f : 0.f;
}

// ---------------- main decoder: 16 (b,k)-rows per block, bf16 WMMA GEMM pair
#define A1S (HIDv + 8)   // 520 halves -> 1040B row stride (16B aligned, bank-staggered)
#define A2S (MIDD + 8)   // 392 halves -> 784B row stride

__global__ __launch_bounds__(256) void k_decoder(
    const float* __restrict__ z, const float* __restrict__ keys,
    const v16bf* __restrict__ W1f, const v16bf* __restrict__ W2f,
    const float* __restrict__ b1, const float* __restrict__ b2,
    const int* __restrict__ nvec, float* __restrict__ out)
{
    __shared__ __bf16 A1[16 * A1S];
    __shared__ __bf16 A2[16 * A2S];

    int tid  = threadIdx.x;
    int wave = tid >> 5, lane = tid & 31;
    int row_base = blockIdx.x * 16;          // rows are (b,k) flattened; 16 rows share one b
    int bb = row_base >> 7;
    int k_base = row_base & (MAXNv - 1);
    int nlim = nvec[bb];

    // A1[m][h] = bf16( z[bb,h] * keys[k_base+m, h] )
    for (int i = tid; i < 16 * HIDv; i += 256) {
        int m = i >> 9, h = i & (HIDv - 1);
        A1[m * A1S + h] = (__bf16)(z[bb * HIDv + h] * keys[(k_base + m) * HIDv + h]);
    }
    __syncthreads();

    int rowA = lane & 15;
    int sel  = (lane < 16) ? 0 : 8;          // ISA 16-bit A layout half-select
    int mrow = (lane < 16) ? 0 : 8;          // C/D layout: M = r + 8*(lane>=16)

    // ---- GEMM1: [16,512]x[512,384], mish epilogue -> A2 (bf16)
    #pragma unroll
    for (int t = 0; t < 3; t++) {
        int nt = wave + t * 8;               // 0..23
        v8f acc = {0.f, 0.f, 0.f, 0.f, 0.f, 0.f, 0.f, 0.f};
        #pragma unroll 4
        for (int kt = 0; kt < 16; kt++) {
            int ko = kt * 32 + sel;
            v8bf a0 = *(const v8bf*)&A1[rowA * A1S + ko];
            v8bf a1 = *(const v8bf*)&A1[rowA * A1S + ko + 16];
            v16bf a = __builtin_shufflevector(a0, a1, 0,1,2,3,4,5,6,7,8,9,10,11,12,13,14,15);
            v16bf bf = W1f[(kt * 24 + nt) * 32 + lane];
            acc = __builtin_amdgcn_wmma_f32_16x16x32_bf16(false, a, false, bf,
                                                          (short)0, acc, false, false);
        }
        int ncol = nt * 16 + (lane & 15);
        float bias = b1[ncol];
        #pragma unroll
        for (int r = 0; r < 8; r++)
            A2[(mrow + r) * A2S + ncol] = (__bf16)mishf(acc[r] + bias);
    }
    __syncthreads();

    // ---- GEMM2: [16,384]x[384,256], bias + ragged mask epilogue -> out (NT stores)
    #pragma unroll
    for (int t = 0; t < 2; t++) {
        int nt = wave + t * 8;               // 0..15
        v8f acc = {0.f, 0.f, 0.f, 0.f, 0.f, 0.f, 0.f, 0.f};
        #pragma unroll 4
        for (int kt = 0; kt < 12; kt++) {
            int ko = kt * 32 + sel;
            v8bf a0 = *(const v8bf*)&A2[rowA * A2S + ko];
            v8bf a1 = *(const v8bf*)&A2[rowA * A2S + ko + 16];
            v16bf a = __builtin_shufflevector(a0, a1, 0,1,2,3,4,5,6,7,8,9,10,11,12,13,14,15);
            v16bf bf = W2f[(kt * 16 + nt) * 32 + lane];
            acc = __builtin_amdgcn_wmma_f32_16x16x32_bf16(false, a, false, bf,
                                                          (short)0, acc, false, false);
        }
        int ncol = nt * 16 + (lane & 15);
        float bias = b2[ncol];
        #pragma unroll
        for (int r = 0; r < 8; r++) {
            int grow = row_base + mrow + r;
            float mv = ((grow & (MAXNv - 1)) < nlim) ? 1.f : 0.f;
            // 256MB streaming output > 192MB L2: non-temporal store keeps the
            // re-read weight/keys working set resident in L2.
            __builtin_nontemporal_store((acc[r] + bias) * mv,
                                        &out[(size_t)grow * DIMv + ncol]);
        }
    }
}

// ---------------- host side ----------------
extern "C" void kernel_launch(void* const* d_in, const int* in_sizes, int n_in,
                              void* d_out, int out_size, void* d_ws, size_t ws_size,
                              hipStream_t stream) {
    (void)in_sizes; (void)n_in; (void)out_size; (void)ws_size;
    const float* z    = (const float*)d_in[0];
    const float* spW1 = (const float*)d_in[1];
    const float* spb1 = (const float*)d_in[2];
    const float* spg  = (const float*)d_in[3];
    const float* spbe = (const float*)d_in[4];
    const float* spW2 = (const float*)d_in[5];
    const float* spb2 = (const float*)d_in[6];
    const float* knW1 = (const float*)d_in[7];
    const float* knb1 = (const float*)d_in[8];
    const float* kng  = (const float*)d_in[9];
    const float* knbe = (const float*)d_in[10];
    const float* knW2 = (const float*)d_in[11];
    const float* knb2 = (const float*)d_in[12];
    const float* deW1 = (const float*)d_in[13];
    const float* deb1 = (const float*)d_in[14];
    const float* deW2 = (const float*)d_in[15];
    const float* deb2 = (const float*)d_in[16];

    // workspace layout (all chunks 32B aligned)
    char*   ws   = (char*)d_ws;
    int*    n_i  = (int*)ws;                                   //   8 KB
    float*  keys = (float*)(ws + 8192);                        //   1 MB (128x512 f32)
    __bf16* W1f  = (__bf16*)(ws + 8192 + 1048576);             // 384 KB (swizzled bf16)
    __bf16* W2f  = (__bf16*)(ws + 8192 + 1048576 + 393216);    // 192 KB

    // output layout: x [2048*128*256] f32, then n [2048], then mask [2048*128]
    float* out_x = (float*)d_out;
    float* out_n = out_x + (size_t)Bsz * MAXNv * DIMv;
    float* out_m = out_n + Bsz;

    k_keys<<<MAXNv, 512, 0, stream>>>(knW1, knb1, kng, knbe, knW2, knb2, keys);
    k_swizzle<<<(196608 + 255) / 256, 256, 0, stream>>>(deW1, W1f, 24, MIDD, 196608);
    k_swizzle<<<(98304  + 255) / 256, 256, 0, stream>>>(deW2, W2f, 16, DIMv,  98304);
    k_size_pred<<<Bsz, 256, 0, stream>>>(z, spW1, spb1, spg, spbe, spW2, spb2,
                                         n_i, out_n, out_m);
    k_decoder<<<(Bsz * MAXNv) / 16, 256, 0, stream>>>(z, keys,
                                                      (const v16bf*)W1f, (const v16bf*)W2f,
                                                      deb1, deb2, n_i, out_x);
}
